// LM_86543591014538
// MI455X (gfx1250) — matrix operand order repaired and checked by reading, not compiled, not run on hardware
//
#include <hip/hip_runtime.h>
#include <math.h>

// ---------------------------------------------------------------------------
// Problem constants (from the reference)
// ---------------------------------------------------------------------------
#define B_   8
#define T_   128
#define D_   4
#define E_   1024
#define L_   2
#define V_   25000
#define CUT0_ 2000
#define CUT1_ 10000
#define HEAD_ 2002          // CUT0 + 2 clusters
#define T0DIM_ 256
#define T1DIM_ 64
#define NT_  (T_ - D_ + 1)  // 125
#define MDEC_ (B_ * NT_)    // 1000
#define MENC_ (B_ * T_)     // 1024
#define MROWS_ (B_ * NT_ * D_) // 4000

// ---------------------------------------------------------------------------
// WMMA types (gfx1250, wave32)
// ---------------------------------------------------------------------------
typedef __attribute__((ext_vector_type(16))) __bf16 v16bf;
typedef __attribute__((ext_vector_type(8)))  float  v8f;
typedef __attribute__((ext_vector_type(2)))  float  f32x2;
typedef __attribute__((ext_vector_type(2)))  __bf16 bf16x2;

union FragBF { v16bf v; unsigned short h[16]; };

// RNE f32->bf16 via hardware conversion (v_cvt_pk_bf16_f32 on gfx1250)
__device__ __forceinline__ unsigned short bf16bits(float f) {
  __bf16 h = (__bf16)f;
  return __builtin_bit_cast(unsigned short, h);
}
__device__ __forceinline__ unsigned pack_bf2(float lo, float hi) {
  f32x2 s; s.x = lo; s.y = hi;
  bf16x2 p = __builtin_convertvector(s, bf16x2);
  return __builtin_bit_cast(unsigned, p);
}

#define TM 128
#define TN 128
#define TKC 32
#define LP (TKC + 8)   // LDS row pitch (ushort): 80B, keeps 16B alignment

// ---------------------------------------------------------------------------
// Fast GEMM:  C[M,N] = A[M,K] * B[N,K]^T (+ bias[col]),  requires K % 32 == 0
// and 16B-aligned rows (lda/ldb multiples of 4) — true for every call here.
// Tile 128x128, double-buffered LDS, software-pipelined staging:
// interior tiles use float4 global loads + ds_store_b128 (no guards);
// edge tiles (block-uniform) use guarded scalar loads.
// 8 wave32 waves; each wave = 32x64 patch = 2x4 v_wmma_f32_16x16x32_bf16.
// ---------------------------------------------------------------------------
__global__ __launch_bounds__(256)
void gemm_nt_wmma(const float* __restrict__ A, const float* __restrict__ B,
                  float* __restrict__ C, const float* __restrict__ bias,
                  int M, int N, int K, int lda, int ldb, int ldc)
{
  __shared__ alignas(16) unsigned short As[2][TM][LP];
  __shared__ alignas(16) unsigned short Bs[2][TN][LP];

  const int tileN = blockIdx.x * TN;
  const int tileM = blockIdx.y * TM;
  const bool edge = (tileM + TM > M) || (tileN + TN > N);

  const int tid  = threadIdx.x;
  const int lane = tid & 31;
  const int w    = tid >> 5;
  const int wm   = (w & 3) * 32;
  const int wn   = (w >> 2) * 64;
  const int lh   = lane & 15;
  const int hi   = lane >> 4;

  // octet tasks: 512 octets of 8 k-elements per tile side, 2 per thread
  const int ar0 = tid >> 2,          ak0 = (tid & 3) * 8;
  const int ar1 = (tid + 256) >> 2,  ak1 = ((tid + 256) & 3) * 8;

  float av[2][8], bv[2][8];

  auto load_side = [&](const float* P, int ld, int lim, int base0, int r0, int k0,
                       int r1, int k1, int kk, float v[2][8]) {
    const int rr[2] = { r0, r1 };
    const int kc[2] = { k0, k1 };
#pragma unroll
    for (int o = 0; o < 2; ++o) {
      long long off = (long long)(base0 + rr[o]) * ld + kk + kc[o];
      if (!edge) {
        float4 x0 = *reinterpret_cast<const float4*>(P + off);
        float4 x1 = *reinterpret_cast<const float4*>(P + off + 4);
        v[o][0]=x0.x; v[o][1]=x0.y; v[o][2]=x0.z; v[o][3]=x0.w;
        v[o][4]=x1.x; v[o][5]=x1.y; v[o][6]=x1.z; v[o][7]=x1.w;
      } else {
        bool inb = (base0 + rr[o]) < lim;
#pragma unroll
        for (int j = 0; j < 8; ++j) v[o][j] = inb ? P[off + j] : 0.f;
      }
    }
  };

  auto store_side = [&](unsigned short (*S)[LP], int r0, int k0, int r1, int k1,
                        float v[2][8]) {
    const int rr[2] = { r0, r1 };
    const int kc[2] = { k0, k1 };
#pragma unroll
    for (int o = 0; o < 2; ++o) {
      uint4 wv;
      wv.x = pack_bf2(v[o][0], v[o][1]);
      wv.y = pack_bf2(v[o][2], v[o][3]);
      wv.z = pack_bf2(v[o][4], v[o][5]);
      wv.w = pack_bf2(v[o][6], v[o][7]);
      *reinterpret_cast<uint4*>(&S[rr[o]][kc[o]]) = wv;
    }
  };

  v8f acc[2][4];
  v8f vz = {0.f,0.f,0.f,0.f,0.f,0.f,0.f,0.f};
#pragma unroll
  for (int i = 0; i < 2; ++i)
#pragma unroll
    for (int j = 0; j < 4; ++j) acc[i][j] = vz;

  const int akb0 = hi ? 8 : 0;
  const int akb1 = hi ? 24 : 16;
  const int bkb  = hi * 16;

  auto compute = [&](int p) {
    FragBF af[2];
#pragma unroll
    for (int i = 0; i < 2; ++i) {
      int r = wm + 16 * i + lh;
#pragma unroll
      for (int j = 0; j < 8; ++j) {
        af[i].h[j]     = As[p][r][akb0 + j];
        af[i].h[8 + j] = As[p][r][akb1 + j];
      }
    }
#pragma unroll
    for (int j = 0; j < 4; ++j) {
      FragBF bf;
      int n = wn + 16 * j + lh;
#pragma unroll
      for (int t = 0; t < 16; ++t) bf.h[t] = Bs[p][n][t < 8 ? akb0 + t : akb1 + (t - 8)];
#pragma unroll
      for (int i = 0; i < 2; ++i)
        acc[i][j] = __builtin_amdgcn_wmma_f32_16x16x32_bf16(
            false, af[i].v, false, bf.v, (short)0, acc[i][j], false, false);
    }
  };

  // ---- pipelined main loop: 1 barrier per K-chunk ----
  const int nk = K / TKC;
  load_side(A, lda, M, tileM, ar0, ak0, ar1, ak1, 0, av);
  load_side(B, ldb, N, tileN, ar0, ak0, ar1, ak1, 0, bv);
  store_side(As[0], ar0, ak0, ar1, ak1, av);
  store_side(Bs[0], ar0, ak0, ar1, ak1, bv);
  __syncthreads();

  int p = 0;
  for (int c = 0; c < nk; ++c) {
    if (c + 1 < nk) {
      load_side(A, lda, M, tileM, ar0, ak0, ar1, ak1, (c + 1) * TKC, av);
      load_side(B, ldb, N, tileN, ar0, ak0, ar1, ak1, (c + 1) * TKC, bv);
    }
    compute(p);
    if (c + 1 < nk) {
      store_side(As[p ^ 1], ar0, ak0, ar1, ak1, av);
      store_side(Bs[p ^ 1], ar0, ak0, ar1, ak1, bv);
    }
    __syncthreads();
    p ^= 1;
  }

  // ---- C store: VGPR r <-> row = r + 8*hi, col = lane&15 ----
#pragma unroll
  for (int i = 0; i < 2; ++i)
#pragma unroll
    for (int j = 0; j < 4; ++j) {
      int col = tileN + wn + 16 * j + lh;
      if (col >= N) continue;
      float badd = bias ? bias[col] : 0.f;
#pragma unroll
      for (int r = 0; r < 8; ++r) {
        int row = tileM + wm + 16 * i + r + 8 * hi;
        if (row < M) C[(long long)row * ldc + col] = acc[i][j][r] + badd;
      }
    }
}

// ---------------------------------------------------------------------------
// Generic batched GEMM (used only for the tiny einsum wgt = G^T @ f).
//   A stored (K,M): element A[k*lda + m] ; B stored (K,N): element B[k*ldb + n]
// Tiles are always full for this use (M=128, N=1024, K=128), K % 32 == 0.
// ---------------------------------------------------------------------------
__global__ __launch_bounds__(256)
void gemm_tt_wmma(const float* __restrict__ A, const float* __restrict__ B,
                  float* __restrict__ C,
                  int M, int N, int K, int lda, int ldb, int ldc,
                  long long sA, long long sB, long long sC)
{
  __shared__ alignas(16) unsigned short As[TM][LP];
  __shared__ alignas(16) unsigned short Bs[TN][LP];

  const long long bz = blockIdx.z;
  A += bz * sA;  B += bz * sB;  C += bz * sC;

  const int tileN = blockIdx.x * TN;
  const int tileM = blockIdx.y * TM;

  const int tid  = threadIdx.x;
  const int lane = tid & 31;
  const int w    = tid >> 5;
  const int wm   = (w & 3) * 32;
  const int wn   = (w >> 2) * 64;
  const int lh   = lane & 15;
  const int hi   = lane >> 4;

  v8f acc[2][4];
  v8f vz = {0.f,0.f,0.f,0.f,0.f,0.f,0.f,0.f};
#pragma unroll
  for (int i = 0; i < 2; ++i)
#pragma unroll
    for (int j = 0; j < 4; ++j) acc[i][j] = vz;

  for (int kk = 0; kk < K; kk += TKC) {
#pragma unroll
    for (int it = 0; it < (TM * TKC) / 256; ++it) {
      int i = tid + it * 256;
      int r = i >> 5, k = i & 31;
      int gm = tileM + r;
      float v = (gm < M) ? A[(long long)(kk + k) * lda + gm] : 0.f;
      As[r][k] = bf16bits(v);
    }
#pragma unroll
    for (int it = 0; it < (TN * TKC) / 256; ++it) {
      int i = tid + it * 256;
      int n = i >> 5, k = i & 31;
      int gn = tileN + n;
      float v = (gn < N) ? B[(long long)(kk + k) * ldb + gn] : 0.f;
      Bs[n][k] = bf16bits(v);
    }
    __syncthreads();

    FragBF af[2];
    const int akb0 = hi ? 8 : 0;
    const int akb1 = hi ? 24 : 16;
#pragma unroll
    for (int i = 0; i < 2; ++i) {
      int r = wm + 16 * i + lh;
#pragma unroll
      for (int j = 0; j < 8; ++j) {
        af[i].h[j]     = As[r][akb0 + j];
        af[i].h[8 + j] = As[r][akb1 + j];
      }
    }
    const int bkb = hi * 16;
#pragma unroll
    for (int j = 0; j < 4; ++j) {
      FragBF bf;
      int n = wn + 16 * j + lh;
#pragma unroll
      for (int t = 0; t < 16; ++t) bf.h[t] = Bs[n][t < 8 ? akb0 + t : akb1 + (t - 8)];
#pragma unroll
      for (int i = 0; i < 2; ++i)
        acc[i][j] = __builtin_amdgcn_wmma_f32_16x16x32_bf16(
            false, af[i].v, false, bf.v, (short)0, acc[i][j], false, false);
    }
    __syncthreads();
  }

#pragma unroll
  for (int i = 0; i < 2; ++i)
#pragma unroll
    for (int j = 0; j < 4; ++j) {
      int col = tileN + wn + 16 * j + lh;
      if (col >= N) continue;
#pragma unroll
      for (int r = 0; r < 8; ++r) {
        int row = tileM + wm + 16 * i + r + 8 * hi;
        if (row < M) C[(long long)row * ldc + col] = acc[i][j][r];
      }
    }
}

// ---------------------------------------------------------------------------
// Elementwise / gather kernels (E_ = 1024 hardcoded; float4 I/O)
// ---------------------------------------------------------------------------
__global__ void gather_embed(const int* __restrict__ x, const float* __restrict__ emb,
                             float* __restrict__ out) {
  int row = blockIdx.x;
  const float4* src = reinterpret_cast<const float4*>(emb + (long long)x[row] * E_);
  float4* dst = reinterpret_cast<float4*>(out + (long long)row * E_);
  for (int e = threadIdx.x; e < E_ / 4; e += blockDim.x) dst[e] = src[e];
}

// xd[d][b*NT + t][e] = (t+d==0) ? emb[x[b,len[b]-1]] : embedded[b][t+d-1]
__global__ void build_xd(const int* __restrict__ x, const int* __restrict__ lengths,
                         const float* __restrict__ embedded, const float* __restrict__ emb,
                         float* __restrict__ xd) {
  int blk = blockIdx.x;               // d*(B*NT) + b*NT + t
  int t = blk % NT_;
  int b = (blk / NT_) % B_;
  int d = blk / (NT_ * B_);
  int s = t + d;
  const float* srcp;
  if (s == 0) srcp = emb + (long long)x[b * T_ + (lengths[b] - 1)] * E_;
  else        srcp = embedded + ((long long)b * T_ + (s - 1)) * E_;
  const float4* src = reinterpret_cast<const float4*>(srcp);
  float4* dst = reinterpret_cast<float4*>(xd + (long long)blk * E_);
  for (int e = threadIdx.x; e < E_ / 4; e += blockDim.x) dst[e] = src[e];
}

__global__ void copy_h0(const float* __restrict__ f, float* __restrict__ h0) {
  int i4 = blockIdx.x * blockDim.x + threadIdx.x;      // over MDEC_*E_/4
  if (i4 >= MDEC_ * (E_ / 4)) return;
  int m  = i4 >> 8;              // / (E_/4)
  int e4 = i4 & 255;
  int b = m / NT_, t = m % NT_;
  reinterpret_cast<float4*>(h0)[i4] =
      reinterpret_cast<const float4*>(f)[((long long)b * T_ + t) * (E_ / 4) + e4];
}

__global__ void gru_gates(const float* __restrict__ gi, const float* __restrict__ gh,
                          const float* __restrict__ h, float* __restrict__ hnext,
                          float* __restrict__ hscat, int scatOff, int M) {
  int idx = blockIdx.x * blockDim.x + threadIdx.x;
  if (idx >= M * E_) return;
  int e = idx & (E_ - 1);
  int m = idx >> 10;             // / E_
  long long base = (long long)m * 3 * E_ + e;
  float r = 1.f / (1.f + __expf(-(gi[base]          + gh[base])));
  float z = 1.f / (1.f + __expf(-(gi[base + E_]     + gh[base + E_])));
  float n = tanhf(gi[base + 2 * E_] + r * gh[base + 2 * E_]);
  float hv = h[(long long)m * E_ + e];
  float o = (1.f - z) * n + z * hv;
  hnext[(long long)m * E_ + e] = o;
  if (hscat) hscat[((long long)m * D_ + scatOff) * E_ + e] = o;
}

// ---------------------------------------------------------------------------
// LDS-resident log-softmax kernels (a 15000-float row = 60KB fits WGP LDS)
// ---------------------------------------------------------------------------
__device__ __forceinline__ float blk_reduce(float v, float* red, bool is_max) {
  int tid = threadIdx.x;
  red[tid] = v;
  __syncthreads();
  for (int s = 128; s > 0; s >>= 1) {
    if (tid < s) red[tid] = is_max ? fmaxf(red[tid], red[tid + s])
                                   : (red[tid] + red[tid + s]);
    __syncthreads();
  }
  float r = red[0];
  __syncthreads();
  return r;
}

__global__ void head_logsoftmax(const float* __restrict__ logits, float* __restrict__ out,
                                float* __restrict__ c0, float* __restrict__ c1,
                                int len, int outld, int outcols) {
  extern __shared__ float smem[];
  float* buf = smem;
  float* red = smem + len;
  int row = blockIdx.x;
  const float* src = logits + (long long)row * len;
  for (int i = threadIdx.x; i < len; i += blockDim.x) buf[i] = src[i];
  __syncthreads();
  float m = -3.0e38f;
  for (int i = threadIdx.x; i < len; i += blockDim.x) m = fmaxf(m, buf[i]);
  m = blk_reduce(m, red, true);
  float s = 0.f;
  for (int i = threadIdx.x; i < len; i += blockDim.x) s += __expf(buf[i] - m);
  s = blk_reduce(s, red, false);
  float lse = m + __logf(s);
  float* dst = out + (long long)row * outld;
  for (int i = threadIdx.x; i < outcols; i += blockDim.x) dst[i] = buf[i] - lse;
  if (threadIdx.x == 0) {
    c0[row] = buf[outcols]     - lse;
    c1[row] = buf[outcols + 1] - lse;
  }
}

// in-place on out[row, coloff : coloff+len), adds cluster log-prob cadd[row].
// len % 4 == 0 and the segment is 16B aligned -> float4 I/O.
__global__ void tail_logsoftmax(float* __restrict__ out, const float* __restrict__ cadd,
                                int len, int outld, int coloff) {
  extern __shared__ float smem[];
  float* buf = smem;
  float* red = smem + len;
  int row = blockIdx.x;
  float4* seg = reinterpret_cast<float4*>(out + (long long)row * outld + coloff);
  float4* b4  = reinterpret_cast<float4*>(buf);
  int len4 = len >> 2;
  for (int i = threadIdx.x; i < len4; i += blockDim.x) b4[i] = seg[i];
  __syncthreads();
  float m = -3.0e38f;
  for (int i = threadIdx.x; i < len; i += blockDim.x) m = fmaxf(m, buf[i]);
  m = blk_reduce(m, red, true);
  float s = 0.f;
  for (int i = threadIdx.x; i < len; i += blockDim.x) s += __expf(buf[i] - m);
  s = blk_reduce(s, red, false);
  float shift = m + __logf(s) - cadd[row];
  for (int i = threadIdx.x; i < len4; i += blockDim.x) {
    float4 v = b4[i];
    v.x -= shift; v.y -= shift; v.z -= shift; v.w -= shift;
    seg[i] = v;
  }
}

// ---------------------------------------------------------------------------
// Host orchestration
// ---------------------------------------------------------------------------
extern "C" void kernel_launch(void* const* d_in, const int* in_sizes, int n_in,
                              void* d_out, int out_size, void* d_ws, size_t ws_size,
                              hipStream_t stream) {
  (void)in_sizes; (void)n_in; (void)out_size; (void)ws_size;

  const int*   x       = (const int*)  d_in[0];
  const int*   lengths = (const int*)  d_in[1];
  const float* emb     = (const float*)d_in[2];
  const float* G       = (const float*)d_in[3];
  const float* eWih    = (const float*)d_in[4];
  const float* eWhh    = (const float*)d_in[5];
  const float* ebih    = (const float*)d_in[6];
  const float* ebhh    = (const float*)d_in[7];
  const float* dWih    = (const float*)d_in[8];
  const float* dWhh    = (const float*)d_in[9];
  const float* dbih    = (const float*)d_in[10];
  const float* dbhh    = (const float*)d_in[11];
  const float* headW   = (const float*)d_in[12];
  const float* t0P     = (const float*)d_in[13];
  const float* t0W     = (const float*)d_in[14];
  const float* t1P     = (const float*)d_in[15];
  const float* t1W     = (const float*)d_in[16];
  float* out = (float*)d_out;

  // ---- workspace carving (floats; every block is a multiple of 4 floats) ----
  float* wsf = (float*)d_ws;
  size_t off = 0;
  auto alloc = [&](size_t n) { float* p = wsf + off; off += n; return p; };
  float* embedded = alloc((size_t)MENC_ * E_);
  float* xd       = alloc((size_t)D_ * MDEC_ * E_);
  float* fA       = alloc((size_t)MENC_ * E_);
  float* fB       = alloc((size_t)MENC_ * E_);
  float* wgt      = alloc((size_t)MENC_ * E_);
  float* gi       = alloc((size_t)MENC_ * 3 * E_);
  float* gh       = alloc((size_t)MENC_ * 3 * E_);
  float* hA       = alloc((size_t)MDEC_ * E_);
  float* hB       = alloc((size_t)MDEC_ * E_);
  float* Hbig     = alloc((size_t)MROWS_ * E_);
  float* headlog  = alloc((size_t)MROWS_ * HEAD_);
  float* c0       = alloc(MROWS_);
  float* c1       = alloc(MROWS_);
  float* h2       = alloc((size_t)MROWS_ * T0DIM_);

  auto gemm_nt = [&](const float* A, const float* Bp, float* Cp, const float* bias,
                     int M, int N, int K, int lda, int ldb, int ldc) {
    dim3 g((N + TN - 1) / TN, (M + TM - 1) / TM, 1);
    gemm_nt_wmma<<<g, 256, 0, stream>>>(A, Bp, Cp, bias, M, N, K, lda, ldb, ldc);
  };

  // ---- embedding gather + decoder input windows ----
  gather_embed<<<MENC_, 256, 0, stream>>>(x, emb, embedded);
  build_xd<<<D_ * MDEC_, 256, 0, stream>>>(x, lengths, embedded, emb, xd);

  // ---- encoder: 2 layers of (G^T @ f) -> GRU cell ----
  const float* fcur = embedded;
  float* fnext = fA;
  for (int l = 0; l < L_; ++l) {
    // wgt[b,t,e] = sum_s G[b,l,s,t] * f[b,s,e]
    dim3 g(E_ / TN, 1, B_);
    gemm_tt_wmma<<<g, 256, 0, stream>>>(G + (size_t)l * T_ * T_, fcur, wgt,
                                        T_, E_, T_, T_, E_, E_,
                                        (long long)L_ * T_ * T_,
                                        (long long)T_ * E_, (long long)T_ * E_);
    gemm_nt(wgt,  eWih + (size_t)l * 3 * E_ * E_, gi, ebih + (size_t)l * 3 * E_,
            MENC_, 3 * E_, E_, E_, E_, 3 * E_);
    gemm_nt(fcur, eWhh + (size_t)l * 3 * E_ * E_, gh, ebhh + (size_t)l * 3 * E_,
            MENC_, 3 * E_, E_, E_, E_, 3 * E_);
    gru_gates<<<(MENC_ * E_ + 255) / 256, 256, 0, stream>>>(
        gi, gh, fcur, fnext, nullptr, 0, MENC_);
    fcur = fnext;
    fnext = (fnext == fA) ? fB : fA;
  }

  // ---- decoder: 4 sequential GRU steps, scatter into Hbig rows (m*D + d) ----
  copy_h0<<<(MDEC_ * (E_ / 4) + 255) / 256, 256, 0, stream>>>(fcur, hA);
  {
    float* hcur = hA;
    float* hnext = hB;
    for (int d = 0; d < D_; ++d) {
      gemm_nt(xd + (size_t)d * MDEC_ * E_, dWih, gi, dbih,
              MDEC_, 3 * E_, E_, E_, E_, 3 * E_);
      gemm_nt(hcur, dWhh, gh, dbhh,
              MDEC_, 3 * E_, E_, E_, E_, 3 * E_);
      gru_gates<<<(MDEC_ * E_ + 255) / 256, 256, 0, stream>>>(
          gi, gh, hcur, hnext, Hbig, d, MDEC_);
      float* t = hcur; hcur = hnext; hnext = t;
    }
  }

  // ---- adaptive softmax head ----
  gemm_nt(Hbig, headW, headlog, nullptr, MROWS_, HEAD_, E_, E_, E_, HEAD_);
  head_logsoftmax<<<MROWS_, 256, (HEAD_ + 256) * sizeof(float), stream>>>(
      headlog, out, c0, c1, HEAD_, V_, CUT0_);

  // ---- tail 0: logits written straight into d_out, then in-place softmax ----
  gemm_nt(Hbig, t0P, h2, nullptr, MROWS_, T0DIM_, E_, E_, E_, T0DIM_);
  gemm_nt(h2, t0W, out + CUT0_, nullptr,
          MROWS_, CUT1_ - CUT0_, T0DIM_, T0DIM_, T0DIM_, V_);
  tail_logsoftmax<<<MROWS_, 256, ((CUT1_ - CUT0_) + 256) * sizeof(float), stream>>>(
      out, c0, CUT1_ - CUT0_, V_, CUT0_);

  // ---- tail 1 ----
  gemm_nt(Hbig, t1P, h2, nullptr, MROWS_, T1DIM_, E_, E_, E_, T1DIM_);
  gemm_nt(h2, t1W, out + CUT1_, nullptr,
          MROWS_, V_ - CUT1_, T1DIM_, T1DIM_, T1DIM_, V_);
  tail_logsoftmax<<<MROWS_, 256, ((V_ - CUT1_) + 256) * sizeof(float), stream>>>(
      out, c1, V_ - CUT1_, V_, CUT1_);
}
